// DecoderBlock_19902878450393
// MI455X (gfx1250) — matrix operand order repaired and checked by reading, not compiled, not run on hardware
//
#include <hip/hip_runtime.h>
#include <hip/hip_bf16.h>

typedef __attribute__((ext_vector_type(16))) __bf16 v16bf;
typedef __attribute__((ext_vector_type(8)))  float  v8f;

#define D_MODEL 512
#define D_STATE 16
#define D_INNER 1024
#define BATCH   4
#define L_TGT   1024
#define L_ENC   1024

// ---------------- helpers ----------------

__device__ __forceinline__ unsigned short f2bf(float f) {
    unsigned int u = __builtin_bit_cast(unsigned int, f);
    unsigned int r = (u + 0x7FFFu + ((u >> 16) & 1u)) >> 16;
    return (unsigned short)r;
}

struct Q2 { uint4 a; uint4 b; };

__device__ __forceinline__ float silu(float v) {
    return v * (1.f / (1.f + __expf(-v)));
}

// LDS byte offset (addrspace(3) numeric value) of a shared-memory pointer.
__device__ __forceinline__ unsigned to_lds(const void* p) {
    return (unsigned)(unsigned long long)
        (__attribute__((address_space(3))) const void*)p;
}

// ---------------- fp32 -> bf16 conversion (with optional K zero-pad) ----------------
// in:  rows x kin  (fp32, dense)
// out: rows x kout (bf16, dense), cols >= kin set to 0

__global__ __launch_bounds__(256)
void f32_to_bf16_kernel(const float* __restrict__ in, unsigned short* __restrict__ out,
                        int rows, int kin, int kout)
{
    int idx = blockIdx.x * 256 + threadIdx.x;
    if (idx >= rows * kout) return;
    int r = idx / kout;
    int c = idx - r * kout;
    float v = (c < kin) ? in[(size_t)r * kin + c] : 0.f;
    out[idx] = f2bf(v);
}

// ---------------- WMMA GEMM (bf16 operands, async-LDS double buffered) ----------------
// C[M,N] = act(A[M,K] @ W[N,K]^T + bias[N]) (+ res[M,N])
// A, W are bf16 (dense strides in elements). M%64==0, N%64==0, K%64==0 required.
// act: 0=none, 1=relu, 2=softplus

#define TM 64
#define TN 64
#define TK 64

__global__ __launch_bounds__(256)
void wmma_gemm_kernel(const unsigned short* __restrict__ A, int lda,
                      const unsigned short* __restrict__ W, int ldw,
                      float* __restrict__ C, int ldc,
                      const float* __restrict__ bias,
                      const float* __restrict__ res, int ldres,
                      int M, int N, int K, int act)
{
    __shared__ __align__(16) unsigned short As[2][TM * TK];
    __shared__ __align__(16) unsigned short Bs[2][TN * TK];

    const int tid  = threadIdx.x;
    const int lane = tid & 31;
    const int w    = tid >> 5;
    const int wm   = w & 3;       // 0..3 -> 16-row slab
    const int wn   = w >> 2;      // 0..1 -> 32-col slab (two 16-col tiles)
    const int rowBase = blockIdx.y * TM;
    const int colBase = blockIdx.x * TN;

    // staging: 64x64 bf16 tile = 512 x 16B chunks; 2 chunks per thread per tile
    const int c0 = tid, c1 = tid + 256;
    const int r0 = c0 >> 3, o0 = (c0 & 7) * 8;
    const int r1 = c1 >> 3, o1 = (c1 & 7) * 8;

    const unsigned asb = to_lds(&As[0][0]);
    const unsigned bsb = to_lds(&Bs[0][0]);

    auto issue = [&](int buf, int k0) {
        unsigned lbuf = (unsigned)(buf * (TM * TK * 2));
        unsigned lA0 = asb + lbuf + (unsigned)((r0 * TK + o0) * 2);
        unsigned lA1 = asb + lbuf + (unsigned)((r1 * TK + o1) * 2);
        unsigned lB0 = bsb + lbuf + (unsigned)((r0 * TK + o0) * 2);
        unsigned lB1 = bsb + lbuf + (unsigned)((r1 * TK + o1) * 2);
        unsigned gA0 = (unsigned)(((size_t)(rowBase + r0) * lda + k0 + o0) * 2);
        unsigned gA1 = (unsigned)(((size_t)(rowBase + r1) * lda + k0 + o1) * 2);
        unsigned gB0 = (unsigned)(((size_t)(colBase + r0) * ldw + k0 + o0) * 2);
        unsigned gB1 = (unsigned)(((size_t)(colBase + r1) * ldw + k0 + o1) * 2);
        asm volatile("global_load_async_to_lds_b128 %0, %1, %2"
                     :: "v"(lA0), "v"(gA0), "s"(A) : "memory");
        asm volatile("global_load_async_to_lds_b128 %0, %1, %2"
                     :: "v"(lA1), "v"(gA1), "s"(A) : "memory");
        asm volatile("global_load_async_to_lds_b128 %0, %1, %2"
                     :: "v"(lB0), "v"(gB0), "s"(W) : "memory");
        asm volatile("global_load_async_to_lds_b128 %0, %1, %2"
                     :: "v"(lB1), "v"(gB1), "s"(W) : "memory");
    };

    v8f acc0 = {};
    v8f acc1 = {};

    issue(0, 0);
    int cur = 0;

    for (int k0 = 0; k0 < K; k0 += TK) {
        if (k0 + TK < K) {
            issue(cur ^ 1, k0 + TK);
            // in-order async completion: <=4 outstanding => current stage landed
            asm volatile("s_wait_asynccnt 0x4" ::: "memory");
        } else {
            asm volatile("s_wait_asynccnt 0x0" ::: "memory");
        }
        __syncthreads();

        const unsigned short* Ab = As[cur];
        const unsigned short* Bb = Bs[cur];
        #pragma unroll
        for (int ks = 0; ks < TK; ks += 32) {
            // A fragment: 16x32, lane = M (mod 16), K contiguous 16 per half-wave
            int m  = wm * 16 + (lane & 15);
            int kh = (lane >> 4) * 16;
            const unsigned short* ab = Ab + m * TK + ks + kh;
            v16bf afrag = __builtin_bit_cast(v16bf,
                Q2{*(const uint4*)(ab), *(const uint4*)(ab + 8)});

            // B fragments: 32x16, lane = N (mod 16), K contiguous 16 per half-wave
            int kb  = (lane >> 4) * 16;
            int nl0 = wn * 32 + (lane & 15);
            const unsigned short* bb0 = Bb + nl0 * TK + ks + kb;
            v16bf bfrag0 = __builtin_bit_cast(v16bf,
                Q2{*(const uint4*)(bb0), *(const uint4*)(bb0 + 8)});
            const unsigned short* bb1 = Bb + (nl0 + 16) * TK + ks + kb;
            v16bf bfrag1 = __builtin_bit_cast(v16bf,
                Q2{*(const uint4*)(bb1), *(const uint4*)(bb1 + 8)});

            acc0 = __builtin_amdgcn_wmma_f32_16x16x32_bf16(
                false, afrag, false, bfrag0, (short)0, acc0, false, false);
            acc1 = __builtin_amdgcn_wmma_f32_16x16x32_bf16(
                false, afrag, false, bfrag1, (short)0, acc1, false, false);
        }
        __syncthreads();
        cur ^= 1;
    }

    // epilogue: C/D layout -> lane n = lane&15, VGPR i -> m = i + 8*(lane>=16)
    int mrow = rowBase + wm * 16 + ((lane >> 4) * 8);
    #pragma unroll
    for (int t = 0; t < 2; ++t) {
        v8f acc = t ? acc1 : acc0;
        int n = colBase + wn * 32 + t * 16 + (lane & 15);
        float bv = bias ? bias[n] : 0.f;
        #pragma unroll
        for (int i = 0; i < 8; ++i) {
            float v = acc[i] + bv;
            if (act == 1) v = fmaxf(v, 0.f);
            else if (act == 2) v = (v > 20.f) ? v : log1pf(__expf(v));
            if (res) v += res[(size_t)(mrow + i) * ldres + n];
            C[(size_t)(mrow + i) * ldc + n] = v;
        }
    }
}

// ---------------- LayerNorm (wave per token, D=512) ----------------

__global__ __launch_bounds__(128)
void ln_kernel(const float* __restrict__ x, const float* __restrict__ g,
               const float* __restrict__ b, float* __restrict__ out,
               int tokens, int l_in, int rpb_out, int row_off)
{
    int lane  = threadIdx.x & 31;
    int token = blockIdx.x * 4 + (threadIdx.x >> 5);
    if (token >= tokens) return;
    const float* xr = x + (size_t)token * D_MODEL;

    float v[16];
    float s = 0.f;
    #pragma unroll
    for (int j = 0; j < 16; ++j) { v[j] = xr[lane + j * 32]; s += v[j]; }
    #pragma unroll
    for (int off = 16; off; off >>= 1) s += __shfl_xor(s, off, 32);
    float mu = s * (1.f / D_MODEL);

    float var = 0.f;
    #pragma unroll
    for (int j = 0; j < 16; ++j) { float d = v[j] - mu; var += d * d; }
    #pragma unroll
    for (int off = 16; off; off >>= 1) var += __shfl_xor(var, off, 32);
    float rs = rsqrtf(var * (1.f / D_MODEL) + 1e-5f);

    int bi = token / l_in, ti = token - bi * l_in;
    float* orow = out + (size_t)(bi * rpb_out + row_off + ti) * D_MODEL;
    #pragma unroll
    for (int j = 0; j < 16; ++j) {
        int d = lane + j * 32;
        orow[d] = (v[j] - mu) * rs * g[d] + b[d];
    }
}

// ---------------- causal depthwise conv (W=4) + SiLU ----------------

__global__ __launch_bounds__(256)
void conv_silu_kernel(const float* __restrict__ xz, int ldxz,
                      const float* __restrict__ cw, const float* __restrict__ cb,
                      float* __restrict__ xi, int total, int L)
{
    int idx = blockIdx.x * 256 + threadIdx.x;
    if (idx >= total) return;
    int d   = idx & (D_INNER - 1);
    int row = idx >> 10;
    int l   = row % L;
    const float* wp = cw + d * 4;
    float acc = cb[d];
    #pragma unroll
    for (int j = 0; j < 4; ++j) {
        int ls = l - 3 + j;
        if (ls >= 0) acc += wp[j] * xz[(size_t)(row - 3 + j) * ldxz + d];
    }
    xi[(size_t)row * D_INNER + d] = silu(acc);
}

// ---------------- selective-scan (state=16) ----------------

__global__ __launch_bounds__(256)
void scan_kernel(const float* __restrict__ delta, const float* __restrict__ dbc,
                 float* __restrict__ xi, const float* __restrict__ xz, int ldxz,
                 const float* __restrict__ A_log, const float* __restrict__ Dp,
                 int L)
{
    __shared__ float Bc[64 * D_STATE];
    __shared__ float Cc[64 * D_STATE];
    const int tid = threadIdx.x;
    const int b   = blockIdx.y;
    const int d   = blockIdx.x * 256 + tid;

    float a2[D_STATE], h[D_STATE];
    #pragma unroll
    for (int n = 0; n < D_STATE; ++n) {
        a2[n] = -__expf(A_log[d * D_STATE + n]) * 1.44269504f;  // A * log2(e)
        h[n]  = 0.f;
    }
    float Dv = Dp[d];
    size_t rowb = (size_t)b * L;

    for (int t0 = 0; t0 < L; t0 += 64) {
        __syncthreads();
        #pragma unroll
        for (int i = 0; i < 8; ++i) {
            int e = i * 256 + tid;              // 0..2047
            int tl = e >> 5, c = e & 31;
            float v = dbc[(rowb + t0 + tl) * 64 + 32 + c];
            if (c < D_STATE) Bc[tl * D_STATE + c] = v;
            else             Cc[tl * D_STATE + (c - D_STATE)] = v;
        }
        __syncthreads();
        for (int tl = 0; tl < 64; ++tl) {
            size_t row = rowb + t0 + tl;
            float dt = delta[row * D_INNER + d];
            float xv = xi[row * D_INNER + d];
            float zv = xz[row * ldxz + D_INNER + d];
            float dBu = dt * xv;
            float y = 0.f;
            #pragma unroll
            for (int n = 0; n < D_STATE; ++n) {
                float dA = exp2f(dt * a2[n]);
                h[n] = dA * h[n] + dBu * Bc[tl * D_STATE + n];
                y += h[n] * Cc[tl * D_STATE + n];
            }
            y += xv * Dv;
            y *= silu(zv);
            xi[row * D_INNER + d] = y;
        }
    }
}

// ---------------- small glue kernels ----------------

__global__ __launch_bounds__(256)
void copy_enc_kernel(const float* __restrict__ enc, float* __restrict__ cat)
{
    int idx = blockIdx.x * 256 + threadIdx.x;       // B*1024*512
    if (idx >= BATCH * L_ENC * D_MODEL) return;
    int d = idx & (D_MODEL - 1);
    int r = idx >> 9;
    int t = r & (L_ENC - 1);
    int b = r >> 10;
    cat[(size_t)(b * (L_ENC + L_TGT) + t) * D_MODEL + d] = enc[idx];
}

__global__ __launch_bounds__(256)
void add_slice_kernel(float* __restrict__ h, const float* __restrict__ cm)
{
    int idx = blockIdx.x * 256 + threadIdx.x;       // B*1024*512
    if (idx >= BATCH * L_TGT * D_MODEL) return;
    int d = idx & (D_MODEL - 1);
    int r = idx >> 9;
    int t = r & (L_TGT - 1);
    int b = r >> 10;
    h[idx] += cm[(size_t)(b * (L_ENC + L_TGT) + L_ENC + t) * D_MODEL + d];
}

// ---------------- host-side orchestration ----------------

struct Ctx {
    unsigned short* bf_a;   // bf16 activation staging
    unsigned short* bf_w;   // bf16 weight staging
    hipStream_t s;
};

static inline void conv_bf16(const float* in, unsigned short* out,
                             int rows, int kin, int kout, hipStream_t s)
{
    int total = rows * kout;
    f32_to_bf16_kernel<<<(total + 255) / 256, 256, 0, s>>>(in, out, rows, kin, kout);
}

// C = act(A_f32[M,K] @ W_f32[N,Kw]^T + bias) (+res); converts both operands to bf16.
// kpad >= K (weight zero-padded to kpad; A garbage cols beyond real K hit zero weights)
static inline void gemm(Ctx& cx, const float* A, int kA,
                        const float* W, int kW, int kpad,
                        float* C, int ldc, const float* bias,
                        const float* res, int ldres,
                        int M, int N, int act)
{
    conv_bf16(A, cx.bf_a, M, kA, kpad, cx.s);
    conv_bf16(W, cx.bf_w, N, kW, kpad, cx.s);
    dim3 grid(N / TN, M / TM);
    wmma_gemm_kernel<<<grid, 256, 0, cx.s>>>(cx.bf_a, kpad, cx.bf_w, kpad,
                                             C, ldc, bias, res, ldres,
                                             M, N, kpad, act);
}

extern "C" void kernel_launch(void* const* d_in, const int* in_sizes, int n_in,
                              void* d_out, int out_size, void* d_ws, size_t ws_size,
                              hipStream_t stream) {
    (void)in_sizes; (void)n_in; (void)ws_size; (void)out_size;

    const float* x       = (const float*)d_in[0];
    const float* enc_out = (const float*)d_in[1];
    const float* ln1_g = (const float*)d_in[2];
    const float* ln1_b = (const float*)d_in[3];
    const float* ln2_g = (const float*)d_in[4];
    const float* ln2_b = (const float*)d_in[5];
    const float* ln3_g = (const float*)d_in[6];
    const float* ln3_b = (const float*)d_in[7];
    const float* ff_w1 = (const float*)d_in[8];
    const float* ff_b1 = (const float*)d_in[9];
    const float* ff_w2 = (const float*)d_in[10];
    const float* ff_b2 = (const float*)d_in[11];

    const float* u_in_w    = (const float*)d_in[12];
    const float* u_conv_w  = (const float*)d_in[13];
    const float* u_conv_b  = (const float*)d_in[14];
    const float* u_xproj_w = (const float*)d_in[15];
    const float* u_dt_w    = (const float*)d_in[16];
    const float* u_dt_b    = (const float*)d_in[17];
    const float* u_A_log   = (const float*)d_in[18];
    const float* u_D       = (const float*)d_in[19];
    const float* u_out_w   = (const float*)d_in[20];

    const float* c_in_w    = (const float*)d_in[21];
    const float* c_conv_w  = (const float*)d_in[22];
    const float* c_conv_b  = (const float*)d_in[23];
    const float* c_xproj_w = (const float*)d_in[24];
    const float* c_dt_w    = (const float*)d_in[25];
    const float* c_dt_b    = (const float*)d_in[26];
    const float* c_A_log   = (const float*)d_in[27];
    const float* c_D       = (const float*)d_in[28];
    const float* c_out_w   = (const float*)d_in[29];

    float* out = (float*)d_out;

    const int M1 = BATCH * L_TGT;             // 4096
    const int M2 = BATCH * (L_ENC + L_TGT);   // 8192

    size_t off = 0;
    auto carve = [&](size_t bytes) {
        void* p = (char*)d_ws + off;
        off += (bytes + 255) & ~(size_t)255;
        return p;
    };
    float* t_cat = (float*)carve((size_t)M2 * D_MODEL * 4);      // ln out / concat / cm out
    float* t_xz  = (float*)carve((size_t)M2 * 2 * D_INNER * 4);  // xz ; reused as ff hidden
    float* t_xi  = (float*)carve((size_t)M2 * D_INNER * 4);      // conv out -> y in place
    float* t_dbc = (float*)carve((size_t)M2 * 64 * 4);           // dt|B|C
    float* t_dlt = (float*)carve((size_t)M2 * D_INNER * 4);      // delta ; reused as ff ln in
    float* t_h   = (float*)carve((size_t)M1 * D_MODEL * 4);      // residual stream

    Ctx cx;
    cx.bf_a = (unsigned short*)carve((size_t)M2 * D_INNER * 2);  // max activation bf16
    cx.bf_w = (unsigned short*)carve((size_t)2 * D_INNER * D_MODEL * 2); // max weight bf16
    cx.s = stream;

    // ===== uni-directional mamba (M1 rows, L=1024) =====
    ln_kernel<<<M1 / 4, 128, 0, stream>>>(x, ln1_g, ln1_b, t_cat, M1, L_TGT, L_TGT, 0);
    gemm(cx, t_cat, D_MODEL, u_in_w, D_MODEL, D_MODEL,
         t_xz, 2 * D_INNER, nullptr, nullptr, 0, M1, 2 * D_INNER, 0);
    {
        int total = M1 * D_INNER;
        conv_silu_kernel<<<(total + 255) / 256, 256, 0, stream>>>(
            t_xz, 2 * D_INNER, u_conv_w, u_conv_b, t_xi, total, L_TGT);
    }
    gemm(cx, t_xi, D_INNER, u_xproj_w, D_INNER, D_INNER,
         t_dbc, 64, nullptr, nullptr, 0, M1, 64, 0);
    gemm(cx, t_dbc, 64, u_dt_w, 32, 64,                      // weight zero-padded 32->64
         t_dlt, D_INNER, u_dt_b, nullptr, 0, M1, D_INNER, 2); // softplus
    {
        dim3 grid(D_INNER / 256, BATCH);
        scan_kernel<<<grid, 256, 0, stream>>>(t_dlt, t_dbc, t_xi, t_xz, 2 * D_INNER,
                                              u_A_log, u_D, L_TGT);
    }
    gemm(cx, t_xi, D_INNER, u_out_w, D_INNER, D_INNER,       // h = x + y @ out_w^T
         t_h, D_MODEL, nullptr, x, D_MODEL, M1, D_MODEL, 0);

    // ===== cross mamba (concat enc_out + ln2(h); M2 rows, L=2048) =====
    {
        int total = BATCH * L_ENC * D_MODEL;
        copy_enc_kernel<<<(total + 255) / 256, 256, 0, stream>>>(enc_out, t_cat);
    }
    ln_kernel<<<M1 / 4, 128, 0, stream>>>(t_h, ln2_g, ln2_b, t_cat, M1, L_TGT,
                                          L_ENC + L_TGT, L_ENC);
    gemm(cx, t_cat, D_MODEL, c_in_w, D_MODEL, D_MODEL,
         t_xz, 2 * D_INNER, nullptr, nullptr, 0, M2, 2 * D_INNER, 0);
    {
        int total = M2 * D_INNER;
        conv_silu_kernel<<<(total + 255) / 256, 256, 0, stream>>>(
            t_xz, 2 * D_INNER, c_conv_w, c_conv_b, t_xi, total, L_ENC + L_TGT);
    }
    gemm(cx, t_xi, D_INNER, c_xproj_w, D_INNER, D_INNER,
         t_dbc, 64, nullptr, nullptr, 0, M2, 64, 0);
    gemm(cx, t_dbc, 64, c_dt_w, 32, 64,
         t_dlt, D_INNER, c_dt_b, nullptr, 0, M2, D_INNER, 2);
    {
        dim3 grid(D_INNER / 256, BATCH);
        scan_kernel<<<grid, 256, 0, stream>>>(t_dlt, t_dbc, t_xi, t_xz, 2 * D_INNER,
                                              c_A_log, c_D, L_ENC + L_TGT);
    }
    gemm(cx, t_xi, D_INNER, c_out_w, D_INNER, D_INNER,
         t_cat, D_MODEL, nullptr, nullptr, 0, M2, D_MODEL, 0);
    {
        int total = BATCH * L_TGT * D_MODEL;
        add_slice_kernel<<<(total + 255) / 256, 256, 0, stream>>>(t_h, t_cat);
    }

    // ===== feed-forward + final residual -> d_out =====
    ln_kernel<<<M1 / 4, 128, 0, stream>>>(t_h, ln3_g, ln3_b, t_dlt, M1, L_TGT, L_TGT, 0);
    gemm(cx, t_dlt, D_MODEL, ff_w1, D_MODEL, D_MODEL,
         t_xz, 2048, ff_b1, nullptr, 0, M1, 2048, 1);        // relu
    gemm(cx, t_xz, 2048, ff_w2, 2048, 2048,
         out, D_MODEL, ff_b2, t_h, D_MODEL, M1, D_MODEL, 0);
}
